// Attention_84817014161988
// MI455X (gfx1250) — matrix operand order repaired and checked by reading, not compiled, not run on hardware
//
#include <hip/hip_runtime.h>
#include <hip/hip_bf16.h>
#include <hip/hip_fp16.h>

typedef __attribute__((ext_vector_type(16))) _Float16 v16h;
typedef __attribute__((ext_vector_type(8)))  float    v8f;

#define N_SEQ  16384
#define DHEAD  64
#define NQ     32     // queries per trunk
#define NK     128    // keys per window
#define PADL   48     // (NK - NQ) / 2

// LDS row strides (in elements), padded to dodge bank conflicts
#define QSTR   72
#define KSTR   72
#define VSTR   136
#define SSTR   129
#define PSTR   132

// A-matrix 16x32 f16 fragment per ISA layout:
// lanes 0-15: row M=lane, K = k0+0..7 then k0+16..23
// lanes 16-31: row M=lane-16, K = k0+8..15 then k0+24..31
__device__ __forceinline__ v16h wmma_load_a(const _Float16* base, int stride,
                                            int row0, int k0, int lane) {
  const _Float16* p = base + (row0 + (lane & 15)) * stride + k0 + ((lane & 16) ? 8 : 0);
  v16h a;
#pragma unroll
  for (int i = 0; i < 8; ++i) { a[i] = p[i]; a[i + 8] = p[i + 16]; }
  return a;
}

// B-matrix 32x16 f16 fragment: lane = column N, K in same order as A;
// lanes 0-15 hold K = k0+0..7 then k0+16..23, lanes 16-31 hold K = k0+8..15 then k0+24..31.
// base is stored column-major for B: element(k, n) = base[(n0+n)*stride + k]
__device__ __forceinline__ v16h wmma_load_b(const _Float16* base, int stride,
                                            int n0, int k0, int lane) {
  const _Float16* p = base + (n0 + (lane & 15)) * stride + k0 + ((lane & 16) ? 8 : 0);
  v16h b;
#pragma unroll
  for (int i = 0; i < 8; ++i) { b[i] = p[i]; b[i + 8] = p[i + 16]; }
  return b;
}

__global__ __launch_bounds__(128)
void local_attn_wmma(const float* __restrict__ Q, const float* __restrict__ K,
                     const float* __restrict__ V, float* __restrict__ O) {
  __shared__ _Float16 Qh[NQ * QSTR];      // Q tile,  [q][d]
  __shared__ _Float16 Kh[NK * KSTR];      // K tile,  [key][d]   (== B col-major for S)
  __shared__ _Float16 Vt[DHEAD * VSTR];   // V tile transposed, [d][key] (== B col-major for O)
  __shared__ float    Sf[NQ * SSTR];      // scores f32
  __shared__ _Float16 Ph[NQ * PSTR];      // exp(S - max) f16
  __shared__ float    rsum[NQ];

  const int tid   = threadIdx.x;
  const int lane  = tid & 31;
  const int wave  = tid >> 5;
  const int trunk = blockIdx.x;
  const int head  = blockIdx.y;
  const int kbase = trunk * NQ - PADL;    // global index of window key 0

  const size_t hbase = (size_t)head * N_SEQ * DHEAD;

  // ---- stage Q tile: 32x64 f32 -> f16 LDS (512 float4, 4 per thread) ----
  {
    const float4* qg = (const float4*)(Q + hbase + (size_t)trunk * NQ * DHEAD);
#pragma unroll
    for (int it = 0; it < 4; ++it) {
      int idx = tid + it * 128;          // 0..511
      int row = idx >> 4, c4 = idx & 15;
      float4 v = qg[idx];
      _Float16* d = &Qh[row * QSTR + c4 * 4];
      d[0] = (_Float16)v.x; d[1] = (_Float16)v.y;
      d[2] = (_Float16)v.z; d[3] = (_Float16)v.w;
    }
  }
  // ---- stage K window (natural) + V window (transposed), zero-fill OOB ----
  {
    const float4* kg = (const float4*)(K + hbase);
    const float4* vg = (const float4*)(V + hbase);
#pragma unroll
    for (int it = 0; it < 16; ++it) {
      int idx = tid + it * 128;          // 0..2047
      int row = idx >> 4, c4 = idx & 15; // key row in window, float4 column
      int kglob = kbase + row;
      float4 kv = {0.f, 0.f, 0.f, 0.f}, vv = {0.f, 0.f, 0.f, 0.f};
      if (kglob >= 0 && kglob < N_SEQ) {
        size_t b = (size_t)kglob * (DHEAD / 4) + c4;
        kv = kg[b];
        vv = vg[b];
      }
      _Float16* dk = &Kh[row * KSTR + c4 * 4];
      dk[0] = (_Float16)kv.x; dk[1] = (_Float16)kv.y;
      dk[2] = (_Float16)kv.z; dk[3] = (_Float16)kv.w;
      int dc = c4 * 4;
      Vt[(dc + 0) * VSTR + row] = (_Float16)vv.x;
      Vt[(dc + 1) * VSTR + row] = (_Float16)vv.y;
      Vt[(dc + 2) * VSTR + row] = (_Float16)vv.z;
      Vt[(dc + 3) * VSTR + row] = (_Float16)vv.w;
    }
  }
  __syncthreads();

  // ---- Phase A: S(32x128) = Q(32x64) . K^T ; 4 tiles/wave, K-dim 64 ----
  {
    int mtile = wave & 1;
#pragma unroll
    for (int i = 0; i < 4; ++i) {
      int ntile = (wave >> 1) * 4 + i;   // 0..7
      v8f acc = {};
#pragma unroll
      for (int kk = 0; kk < DHEAD; kk += 32) {
        v16h a = wmma_load_a(Qh, QSTR, mtile * 16, kk, lane);
        v16h b = wmma_load_b(Kh, KSTR, ntile * 16, kk, lane);
        acc = __builtin_amdgcn_wmma_f32_16x16x32_f16(false, a, false, b,
                                                     (short)0, acc, false, false);
      }
      // D layout: vgpr r -> M = r (+8 for lanes 16-31), N = lane&15
      int n  = ntile * 16 + (lane & 15);
      int mb = mtile * 16 + ((lane & 16) ? 8 : 0);
#pragma unroll
      for (int r = 0; r < 8; ++r) Sf[(mb + r) * SSTR + n] = acc[r];
    }
  }
  __syncthreads();

  // ---- masked softmax: one thread per query row (rows always have valid keys) ----
  if (tid < NQ) {
    const float* srow = &Sf[tid * SSTR];
    int jlo = (kbase < 0) ? -kbase : 0;
    int jhi = (kbase + NK > N_SEQ) ? (N_SEQ - kbase) : NK;
    float mx = -3.0e38f;
    for (int j = jlo; j < jhi; ++j) { float x = srow[j]; mx = (x > mx) ? x : mx; }
    float sum = 0.f;
    _Float16* prow = &Ph[tid * PSTR];
    for (int j = 0; j < NK; ++j) {
      float e = (j >= jlo && j < jhi) ? __expf(srow[j] - mx) : 0.f;
      sum += e;
      prow[j] = (_Float16)e;
    }
    rsum[tid] = sum;
  }
  __syncthreads();

  // ---- Phase C: O(32x64) = P(32x128) . V ; 2 tiles/wave, K-dim 128 ----
  {
    int mtile = wave & 1;
    float* og = O + hbase + (size_t)trunk * NQ * DHEAD;
#pragma unroll
    for (int i = 0; i < 2; ++i) {
      int ntile = (wave >> 1) * 2 + i;   // 0..3
      v8f acc = {};
#pragma unroll
      for (int kk = 0; kk < NK; kk += 32) {
        v16h a = wmma_load_a(Ph, PSTR, mtile * 16, kk, lane);
        v16h b = wmma_load_b(Vt, VSTR, ntile * 16, kk, lane);
        acc = __builtin_amdgcn_wmma_f32_16x16x32_f16(false, a, false, b,
                                                     (short)0, acc, false, false);
      }
      int n  = ntile * 16 + (lane & 15);
      int mb = mtile * 16 + ((lane & 16) ? 8 : 0);
#pragma unroll
      for (int r = 0; r < 8; ++r) {
        int m = mb + r;
        og[m * DHEAD + n] = acc[r] / rsum[m];
      }
    }
  }
}

extern "C" void kernel_launch(void* const* d_in, const int* in_sizes, int n_in,
                              void* d_out, int out_size, void* d_ws, size_t ws_size,
                              hipStream_t stream) {
  const float* q = (const float*)d_in[0];
  const float* k = (const float*)d_in[1];
  const float* v = (const float*)d_in[2];
  float* o = (float*)d_out;
  dim3 grid(N_SEQ / NQ, 16, 1);   // 512 trunks x 16 heads = 8192 blocks
  dim3 block(128, 1, 1);          // 4 waves (wave32)
  local_attn_wmma<<<grid, block, 0, stream>>>(q, k, v, o);
}
